// SpatialGraphConvModule_22239340658858
// MI455X (gfx1250) — matrix operand order repaired and checked by reading, not compiled, not run on hardware
//
#include <hip/hip_runtime.h>

typedef __bf16 bf16;
typedef __attribute__((ext_vector_type(16))) __bf16 v16bf;
typedef __attribute__((ext_vector_type(8)))  __bf16 v8bf;
typedef __attribute__((ext_vector_type(8)))  float   v8f;
typedef __attribute__((ext_vector_type(4)))  unsigned int v4u;
typedef __attribute__((ext_vector_type(8)))  int     v8i;
typedef __attribute__((ext_vector_type(4)))  int     v4i;

#define NSP 4096   // spatial size D*H*W
#define CCH 64     // channels

#if __has_builtin(__builtin_amdgcn_tensor_load_to_lds) && \
    __has_builtin(__builtin_amdgcn_s_wait_tensorcnt)
#define USE_TDM 1
#else
#define USE_TDM 0
#endif

// ---------------- workspace layout (bytes) ----------------
static constexpr size_t SZ_MAT_BF = (size_t)NSP * CCH * 2;            // 512 KB
static constexpr size_t OFF_XT_I = 0;                                  // [n][c] bf16
static constexpr size_t OFF_XT_J = OFF_XT_I + SZ_MAT_BF;
static constexpr size_t OFF_WBF  = OFF_XT_J + SZ_MAT_BF;               // 5x[o][c] bf16
static constexpr size_t OFF_QS_I = OFF_WBF + 5 * 64 * 64 * 2;          // [m][c] bf16 (pre-scaled)
static constexpr size_t OFF_QS_J = OFF_QS_I + SZ_MAT_BF;
static constexpr size_t OFF_KT_I = OFF_QS_J + SZ_MAT_BF;               // [n][c] bf16
static constexpr size_t OFF_KT_J = OFF_KT_I + SZ_MAT_BF;
static constexpr size_t OFF_V_I  = OFF_KT_J + SZ_MAT_BF;               // [c][m] bf16
static constexpr size_t OFF_V_J  = OFF_V_I + SZ_MAT_BF;
static constexpr size_t OFF_ST_I = OFF_V_J + SZ_MAT_BF;                // [n][c] bf16 (msg + residual)
static constexpr size_t OFF_ST_J = OFF_ST_I + SZ_MAT_BF;
static constexpr size_t OFF_RMAX = OFF_ST_J + SZ_MAT_BF;               // [4][4096] f32
static constexpr size_t OFF_RINV = OFF_RMAX + 4 * NSP * 4;             // [4][4096] f32 (1/sumexp)
static constexpr size_t OFF_Y_I  = OFF_RINV + 4 * NSP * 4;             // [c][n] f32
static constexpr size_t OFF_Y_J  = OFF_Y_I + (size_t)CCH * NSP * 4;
static constexpr size_t OFF_BN   = OFF_Y_J + (size_t)CCH * NSP * 4;    // [2][2][64] f32

#define WS(T, off) ((T*)(ws + (off)))

// ---------------- WMMA helpers ----------------
__device__ __forceinline__ v8f wmma_bf(v16bf a, v16bf b, v8f c) {
    return __builtin_amdgcn_wmma_f32_16x16x32_bf16(false, a, false, b, (short)0, c,
                                                   false, false);
}

// A-operand 16x32 bf16: base row-major [row][ld halves], contraction along K.
// lane l<16: row l, K in {kk+0..15}; lane>=16: row l-16, K {kk+16..31}
__device__ __forceinline__ v16bf load_a_tile(const bf16* base, int ld, int row0,
                                             int kk, int lane) {
    const bf16* p = base + (size_t)(row0 + (lane & 15)) * ld + kk + (lane >> 4) * 16;
    v8bf s0 = *(const v8bf*)p;
    v8bf s1 = *(const v8bf*)(p + 8);
    v16bf a;
#pragma unroll
    for (int i = 0; i < 8; ++i) { a[i] = s0[i]; a[i + 8] = s1[i]; }
    return a;
}

// B-operand 32x16 bf16 from memory stored [n][k] (n-major, k contiguous).
// lane l: col l%16, 16 contiguous halves at kk + (l/16)*16
__device__ __forceinline__ v16bf load_b_tile(const bf16* base, int ld, int col0,
                                             int kk, int lane) {
    const bf16* p = base + (size_t)(col0 + (lane & 15)) * ld + kk + (lane >> 4) * 16;
    return *(const v16bf*)p;
}

// ---------------- K0: convert/transpose inputs, convert weights, zero BN sums ----
__global__ void k_prep(char* __restrict__ ws, const float* __restrict__ Xi,
                       const float* __restrict__ Xj, const float* __restrict__ Wq,
                       const float* __restrict__ Wk, const float* __restrict__ Wv,
                       const float* __restrict__ Wpi, const float* __restrict__ Wpj) {
    const int NX = 2 * CCH * NSP;       // 524288
    const int NW = 5 * 64 * 64;         // 20480
    const int total = NX + NW + 256;
    for (int idx = blockIdx.x * blockDim.x + threadIdx.x; idx < total;
         idx += gridDim.x * blockDim.x) {
        if (idx < NX) {
            int br = idx >> 18;
            int rem = idx & ((1 << 18) - 1);
            int c = rem >> 12, n = rem & (NSP - 1);
            const float* X = br ? Xj : Xi;
            bf16* Xt = WS(bf16, br ? OFF_XT_J : OFF_XT_I);
            Xt[(size_t)n * CCH + c] = (bf16)X[rem];
        } else if (idx < NX + NW) {
            int w = idx - NX;
            int wi = w >> 12, e = w & 4095;
            const float* Wsrc = (wi == 0) ? Wq : (wi == 1) ? Wk
                              : (wi == 2) ? Wv : (wi == 3) ? Wpi : Wpj;
            WS(bf16, OFF_WBF)[w] = (bf16)Wsrc[e];
        } else {
            WS(float, OFF_BN)[idx - NX - NW] = 0.0f;
        }
    }
}

// ---------------- K1a: Q/K projection with transposed [m][c] output ------------
__global__ void k_qk_proj(char* __restrict__ ws) {
    int lane = threadIdx.x & 31, wave = threadIdx.x >> 5;
    int tile = blockIdx.x * 8 + wave;
    int m0 = (tile >> 2) << 4;
    int o0 = (tile & 3) << 4;
    int which = blockIdx.y;                        // 0:Qi 1:Ki 2:Qj 3:Kj
    const bf16* Xt = WS(bf16, (which >= 2) ? OFF_XT_J : OFF_XT_I);
    const bf16* W  = WS(bf16, OFF_WBF) + (size_t)(which & 1) * 4096;
    bf16* out = WS(bf16, (which == 0) ? OFF_QS_I : (which == 1) ? OFF_KT_I
                       : (which == 2) ? OFF_QS_J : OFF_KT_J);
    float scale = (which & 1) ? 1.0f : 0.125f;     // 1/sqrt(64) folded into Q
    v8f acc = {};
    acc = wmma_bf(load_a_tile(Xt, CCH, m0, 0, lane),  load_b_tile(W, CCH, o0, 0, lane),  acc);
    acc = wmma_bf(load_a_tile(Xt, CCH, m0, 32, lane), load_b_tile(W, CCH, o0, 32, lane), acc);
    int nn = lane & 15, hl = lane >> 4;
#pragma unroll
    for (int r = 0; r < 8; ++r)
        out[(size_t)(m0 + r + 8 * hl) * CCH + o0 + nn] = (bf16)(acc[r] * scale);
}

// ---------------- K1b: V projection, output [c][m] --------------------------
__global__ void k_v_proj(char* __restrict__ ws) {
    int lane = threadIdx.x & 31, wave = threadIdx.x >> 5;
    int tile = blockIdx.x * 8 + wave;
    int o0 = (tile & 3) << 4;
    int m0 = (tile >> 2) << 4;
    int br = blockIdx.y;
    const bf16* Wv = WS(bf16, OFF_WBF) + 2 * 4096;
    const bf16* Xt = WS(bf16, br ? OFF_XT_J : OFF_XT_I);
    bf16* V = WS(bf16, br ? OFF_V_J : OFF_V_I);
    v8f acc = {};
    acc = wmma_bf(load_a_tile(Wv, CCH, o0, 0, lane),  load_b_tile(Xt, CCH, m0, 0, lane),  acc);
    acc = wmma_bf(load_a_tile(Wv, CCH, o0, 32, lane), load_b_tile(Xt, CCH, m0, 32, lane), acc);
    int nn = lane & 15, hl = lane >> 4;
#pragma unroll
    for (int r = 0; r < 8; ++r)
        V[(size_t)(o0 + r + 8 * hl) * NSP + m0 + nn] = (bf16)acc[r];
}

// ---------------- K2: per-row softmax stats (online max/sumexp) -------------
// pairs: 0:QiKi 1:QiKj 2:QjKj 3:QjKi ; logits already scaled (Q pre-scaled)
__global__ void k_rowstats(char* __restrict__ ws) {
    int lane = threadIdx.x & 31, wave = threadIdx.x >> 5;
    int p = blockIdx.y;
    int m0 = (blockIdx.x * 8 + wave) << 4;
    const bf16* Q = WS(bf16, (p < 2) ? OFF_QS_I : OFF_QS_J);
    const bf16* K = WS(bf16, (p == 0 || p == 3) ? OFF_KT_I : OFF_KT_J);
    v16bf a0 = load_a_tile(Q, CCH, m0, 0, lane);
    v16bf a1 = load_a_tile(Q, CCH, m0, 32, lane);
    float rmax[8], rsum[8];
#pragma unroll
    for (int r = 0; r < 8; ++r) { rmax[r] = -__builtin_inff(); rsum[r] = 0.0f; }
    for (int n0 = 0; n0 < NSP; n0 += 16) {
        if (n0 + 128 < NSP)
            __builtin_prefetch(K + (size_t)(n0 + 128) * CCH, 0, 0);
        v8f acc = {};
        acc = wmma_bf(a0, load_b_tile(K, CCH, n0, 0, lane),  acc);
        acc = wmma_bf(a1, load_b_tile(K, CCH, n0, 32, lane), acc);
#pragma unroll
        for (int r = 0; r < 8; ++r) {
            float x = acc[r];
            float mo = rmax[r];
            float mn = fmaxf(mo, x);
            rsum[r] = rsum[r] * __expf(mo - mn) + __expf(x - mn);
            rmax[r] = mn;
        }
    }
#pragma unroll
    for (int off = 1; off < 16; off <<= 1) {
#pragma unroll
        for (int r = 0; r < 8; ++r) {
            float om = __shfl_xor(rmax[r], off, 16);
            float os = __shfl_xor(rsum[r], off, 16);
            float mn = fmaxf(rmax[r], om);
            rsum[r] = rsum[r] * __expf(rmax[r] - mn) + os * __expf(om - mn);
            rmax[r] = mn;
        }
    }
    if ((lane & 15) == 0) {
        float* RM = WS(float, OFF_RMAX) + (size_t)p * NSP;
        float* RI = WS(float, OFF_RINV) + (size_t)p * NSP;
        int hl = lane >> 4;
#pragma unroll
        for (int r = 0; r < 8; ++r) {
            RM[m0 + r + 8 * hl] = rmax[r];
            RI[m0 + r + 8 * hl] = 1.0f / rsum[r];
        }
    }
}

// ---------------- K3: fused message passing + residual ----------------------
// S[c][n] = sum_m V[c][m] * (P_a + P_b)[m][n] ; St[n][c] = bf16(S + X)
// V is staged in 64x256 chunks into LDS by the Tensor Data Mover (one DMA per
// block instead of 8 redundant L2 streams), consumed as WMMA A-operands via DS.
#define VCHUNK 256
__global__ void k_message(char* __restrict__ ws) {
    __shared__ __align__(64) bf16 vlds[CCH][VCHUNK];   // 32 KB TDM-staged V chunk
    __shared__ __align__(64) bf16 plds[8][16][32];     // per-wave P tiles (8 KB)
    int lane = threadIdx.x & 31, wave = threadIdx.x >> 5;
    int br = blockIdx.y;
    int n0 = (blockIdx.x * 8 + wave) << 4;
    int pa = br ? 2 : 0, pb = br ? 3 : 1;
    const bf16* Qb = WS(bf16, br ? OFF_QS_J : OFF_QS_I);
    const bf16* Ka = WS(bf16, br ? OFF_KT_J : OFF_KT_I);  // same-branch K
    const bf16* Kb = WS(bf16, br ? OFF_KT_I : OFF_KT_J);  // cross-branch K
    const bf16* Vg = WS(bf16, br ? OFF_V_J : OFF_V_I);
    const float* rmaxA = WS(float, OFF_RMAX) + (size_t)pa * NSP;
    const float* rinvA = WS(float, OFF_RINV) + (size_t)pa * NSP;
    const float* rmaxB = WS(float, OFF_RMAX) + (size_t)pb * NSP;
    const float* rinvB = WS(float, OFF_RINV) + (size_t)pb * NSP;

    int nn = lane & 15, hl = lane >> 4;
    v16bf bKa0 = load_b_tile(Ka, CCH, n0, 0, lane);
    v16bf bKa1 = load_b_tile(Ka, CCH, n0, 32, lane);
    v16bf bKb0 = load_b_tile(Kb, CCH, n0, 0, lane);
    v16bf bKb1 = load_b_tile(Kb, CCH, n0, 32, lane);

    v8f acc[4] = {};
    for (int mc = 0; mc < NSP; mc += VCHUNK) {
        __syncthreads();   // previous chunk fully consumed before overwrite
        if (wave == 0) {
#if USE_TDM
            // Tensor DMA descriptor: 2-D tensor, 2-byte elems, tile 256x64
            unsigned long long ga =
                (unsigned long long)(size_t)Vg + (unsigned long long)mc * 2u;
            v4u g0 = { 1u,                                   // count=1
                       (unsigned)(size_t)&vlds[0][0],        // lds_addr
                       (unsigned)(ga & 0xffffffffu),
                       (unsigned)((ga >> 32) & 0x1ffffffu) | 0x80000000u }; // type=2
            v8i g1 = { (int)0x10000,            // data_size = 2 bytes
                       (int)(4096u << 16),      // tensor_dim0 = 4096 (bits 79:48)
                       (int)(64u << 16),        // tensor_dim1 = 64
                       (int)((unsigned)VCHUNK << 16),  // tile_dim0 = 256
                       (int)64,                 // tile_dim1 = 64, tile_dim2 = 0
                       (int)4096,               // tensor_dim0_stride = 4096
                       0, 0 };
            v4i g2 = { 0, 0, 0, 0 };
            v4i g3 = { 0, 0, 0, 0 };
#if __clang_major__ >= 23
            v8i g4 = { 0, 0, 0, 0, 0, 0, 0, 0 };
            __builtin_amdgcn_tensor_load_to_lds(g0, g1, g2, g3, g4, 0);
#else
            __builtin_amdgcn_tensor_load_to_lds(g0, g1, g2, g3, 0);
#endif
            __builtin_amdgcn_s_wait_tensorcnt((short)0);
#else
            // fallback: cooperative copy by wave 0
            for (int t = lane; t < (CCH * VCHUNK) / 8; t += 32) {
                int c = t >> 5, off = (t & 31) * 8;
                *(v8bf*)&vlds[c][off] =
                    *(const v8bf*)(Vg + (size_t)c * NSP + mc + off);
            }
#endif
        }
        __syncthreads();   // staged V visible to all waves

        for (int m0 = mc; m0 < mc + VCHUNK; m0 += 32) {
            if (m0 + 128 < NSP)
                __builtin_prefetch(Qb + (size_t)(m0 + 128) * CCH, 0, 0);
#pragma unroll
            for (int t = 0; t < 2; ++t) {
                int mt = m0 + t * 16;
                v8f la = {}, lb = {};
                la = wmma_bf(load_a_tile(Qb, CCH, mt, 0, lane),  bKa0, la);
                la = wmma_bf(load_a_tile(Qb, CCH, mt, 32, lane), bKa1, la);
                lb = wmma_bf(load_a_tile(Qb, CCH, mt, 0, lane),  bKb0, lb);
                lb = wmma_bf(load_a_tile(Qb, CCH, mt, 32, lane), bKb1, lb);
#pragma unroll
                for (int r = 0; r < 8; ++r) {
                    int m = mt + r + 8 * hl;
                    float p1 = __expf(la[r] - rmaxA[m]) * rinvA[m];
                    float p2 = __expf(lb[r] - rmaxB[m]) * rinvB[m];
                    plds[wave][nn][t * 16 + r + 8 * hl] = (bf16)(p1 + p2);
                }
            }
            // D-layout -> B-operand layout via LDS (single wave: DS in order)
            v16bf pbv = *(const v16bf*)&plds[wave][nn][hl * 16];
            const bf16* Vl = &vlds[0][0];
#pragma unroll
            for (int ct = 0; ct < 4; ++ct)
                acc[ct] = wmma_bf(load_a_tile(Vl, VCHUNK, ct * 16, m0 - mc, lane),
                                  pbv, acc[ct]);
        }
    }
    const bf16* Xt = WS(bf16, br ? OFF_XT_J : OFF_XT_I);
    bf16* St = WS(bf16, br ? OFF_ST_J : OFF_ST_I);
#pragma unroll
    for (int ct = 0; ct < 4; ++ct)
#pragma unroll
        for (int r = 0; r < 8; ++r) {
            size_t addr = (size_t)(n0 + nn) * CCH + ct * 16 + r + 8 * hl;
            St[addr] = (bf16)(acc[ct][r] + (float)Xt[addr]);
        }
}

// ---------------- K4: final 1x1 conv + BN partial sums ----------------------
__global__ void k_final_conv(char* __restrict__ ws) {
    int lane = threadIdx.x & 31, wave = threadIdx.x >> 5;
    int tile = blockIdx.x * 8 + wave;
    int o0 = (tile & 3) << 4;
    int n0 = (tile >> 2) << 4;
    int br = blockIdx.y;
    const bf16* Wp = WS(bf16, OFF_WBF) + (size_t)(3 + br) * 4096;
    const bf16* St = WS(bf16, br ? OFF_ST_J : OFF_ST_I);
    float* Y = WS(float, br ? OFF_Y_J : OFF_Y_I);
    float* bn = WS(float, OFF_BN) + (size_t)br * 128;
    v8f acc = {};
    acc = wmma_bf(load_a_tile(Wp, CCH, o0, 0, lane),  load_b_tile(St, CCH, n0, 0, lane),  acc);
    acc = wmma_bf(load_a_tile(Wp, CCH, o0, 32, lane), load_b_tile(St, CCH, n0, 32, lane), acc);
    int nn = lane & 15, hl = lane >> 4;
#pragma unroll
    for (int r = 0; r < 8; ++r) {
        float y = acc[r];
        Y[(size_t)(o0 + r + 8 * hl) * NSP + n0 + nn] = y;
        float s = y, q = y * y;
#pragma unroll
        for (int off = 1; off < 16; off <<= 1) {
            s += __shfl_xor(s, off, 16);
            q += __shfl_xor(q, off, 16);
        }
        if (nn == 0) {
            atomicAdd(&bn[o0 + r + 8 * hl], s);
            atomicAdd(&bn[64 + o0 + r + 8 * hl], q);
        }
    }
}

// ---------------- K5: BatchNorm + LeakyReLU + branch sum --------------------
__global__ void k_bn_out(char* __restrict__ ws, const float* __restrict__ gi,
                         const float* __restrict__ bi, const float* __restrict__ gj,
                         const float* __restrict__ bj, float* __restrict__ out) {
    const float inv_n = 1.0f / (float)NSP;
    const float* Yi = WS(float, OFF_Y_I);
    const float* Yj = WS(float, OFF_Y_J);
    const float* bn = WS(float, OFF_BN);
    for (int idx = blockIdx.x * blockDim.x + threadIdx.x; idx < CCH * NSP;
         idx += gridDim.x * blockDim.x) {
        int c = idx >> 12;
        float mi = bn[c] * inv_n;
        float vi = bn[64 + c] * inv_n - mi * mi;
        float mj = bn[128 + c] * inv_n;
        float vj = bn[192 + c] * inv_n - mj * mj;
        float yi = (Yi[idx] - mi) * rsqrtf(vi + 1e-5f) * gi[c] + bi[c];
        yi = (yi >= 0.0f) ? yi : 0.01f * yi;
        float yj = (Yj[idx] - mj) * rsqrtf(vj + 1e-5f) * gj[c] + bj[c];
        yj = (yj >= 0.0f) ? yj : 0.01f * yj;
        out[idx] = yi + yj;
    }
}

// ---------------- host launcher ---------------------------------------------
extern "C" void kernel_launch(void* const* d_in, const int* in_sizes, int n_in,
                              void* d_out, int out_size, void* d_ws, size_t ws_size,
                              hipStream_t stream) {
    (void)in_sizes; (void)n_in; (void)out_size; (void)ws_size;
    char* ws = (char*)d_ws;
    const float* Xi  = (const float*)d_in[0];
    const float* Xj  = (const float*)d_in[1];
    const float* Wq  = (const float*)d_in[2];
    const float* Wk  = (const float*)d_in[3];
    const float* Wv  = (const float*)d_in[4];
    const float* Wpi = (const float*)d_in[5];
    const float* Wpj = (const float*)d_in[6];
    const float* gi  = (const float*)d_in[7];
    const float* bi  = (const float*)d_in[8];
    const float* gj  = (const float*)d_in[9];
    const float* bj  = (const float*)d_in[10];
    float* out = (float*)d_out;

    k_prep<<<dim3(512), 256, 0, stream>>>(ws, Xi, Xj, Wq, Wk, Wv, Wpi, Wpj);
    k_qk_proj<<<dim3(128, 4), 256, 0, stream>>>(ws);
    k_v_proj<<<dim3(128, 2), 256, 0, stream>>>(ws);
    k_rowstats<<<dim3(32, 4), 256, 0, stream>>>(ws);
    k_message<<<dim3(32, 2), 256, 0, stream>>>(ws);
    k_final_conv<<<dim3(128, 2), 256, 0, stream>>>(ws);
    k_bn_out<<<dim3(512), 256, 0, stream>>>(ws, gi, bi, gj, bj, out);
}
